// UserGraphLayer_23630910063008
// MI455X (gfx1250) — matrix operand order repaired and checked by reading, not compiled, not run on hardware
//
#include <hip/hip_runtime.h>
#include <cstdint>
#include <cstddef>

// ---------------------------------------------------------------------------
// Problem constants (from reference setup_inputs)
// ---------------------------------------------------------------------------
constexpr int U_N  = 10000;
constexpr int P_N  = 50000;
constexpr int T_N  = 48;
constexpr int W_N  = 10;
constexpr int S_N  = 4;
constexpr int DW_N = 400;
constexpr int MW_N = 12;
constexpr int E_N  = 1000000;
constexpr int DD   = 128;
constexpr float EPSF = 1e-9f;

// ---------------------------------------------------------------------------
// Workspace layout (float offsets), every section aligned to 64 floats (256 B)
// so b128 vector loads of any section are aligned. Accumulators
// [0, WS_ACC_END) are zeroed every launch (harness poisons ws).
// ---------------------------------------------------------------------------
constexpr size_t alignup(size_t x) { return (x + 63) & ~(size_t)63; }

constexpr size_t WS_USER_SUM = 0;
constexpr size_t WS_USER_CNT = alignup(WS_USER_SUM + (size_t)U_N * DD);
constexpr size_t WS_POI_SUM  = alignup(WS_USER_CNT + U_N);
constexpr size_t WS_POI_CNT  = alignup(WS_POI_SUM + (size_t)P_N * DD);
constexpr size_t WS_TIME_SUM = alignup(WS_POI_CNT + P_N);
constexpr size_t WS_TIME_CNT = alignup(WS_TIME_SUM + (size_t)T_N * DD);
constexpr size_t WS_WEA_SUM  = alignup(WS_TIME_CNT + T_N);
constexpr size_t WS_WEA_CNT  = alignup(WS_WEA_SUM + (size_t)W_N * DD);
constexpr size_t WS_SEA_SUM  = alignup(WS_WEA_CNT + W_N);
constexpr size_t WS_SEA_CNT  = alignup(WS_SEA_SUM + (size_t)S_N * DD);
constexpr size_t WS_DAY_SUM  = alignup(WS_SEA_CNT + S_N);
constexpr size_t WS_DAY_CNT  = alignup(WS_DAY_SUM + (size_t)DW_N * DD);
constexpr size_t WS_MON_SUM  = alignup(WS_DAY_CNT + DW_N);
constexpr size_t WS_MON_CNT  = alignup(WS_MON_SUM + (size_t)MW_N * DD);
constexpr size_t WS_ACC_END  = alignup(WS_MON_CNT + MW_N);
constexpr size_t WS_NU       = WS_ACC_END;                        // f32 user means
constexpr size_t WS_SQ       = alignup(WS_NU + (size_t)U_N * DD); // row sq norms
constexpr size_t WS_NUB      = alignup(WS_SQ + U_N);              // bf16 copy (u16)

// Output layout (floats), concatenated in reference return order
constexpr size_t O_USER = 0;
constexpr size_t O_POI  = O_USER + (size_t)U_N * DD;
constexpr size_t O_TIME = O_POI + (size_t)P_N * DD;
constexpr size_t O_WEA  = O_TIME + (size_t)T_N * DD;
constexpr size_t O_SEA  = O_WEA + (size_t)W_N * DD;
constexpr size_t O_DAY  = O_SEA + (size_t)S_N * DD;
constexpr size_t O_MON  = O_DAY + (size_t)DW_N * DD;

// ---------------------------------------------------------------------------
// Vector types
// ---------------------------------------------------------------------------
typedef __attribute__((ext_vector_type(16))) __bf16 v16bf;
typedef __attribute__((ext_vector_type(8)))  float  v8f;
typedef __attribute__((ext_vector_type(4)))  float  f4;
typedef __attribute__((ext_vector_type(4)))  unsigned u4;

// ---------------------------------------------------------------------------
// Kernel 1: zero accumulators
// ---------------------------------------------------------------------------
__global__ void zero_kernel(float* __restrict__ p, size_t n) {
  size_t i = (size_t)blockIdx.x * blockDim.x + threadIdx.x;
  size_t stride = (size_t)gridDim.x * blockDim.x;
  for (; i < n; i += stride) p[i] = 0.0f;
}

// ---------------------------------------------------------------------------
// Kernel 2: edge scatter. One wave per edge slot, lane handles 4 dims.
// Streamed-once data (e_W, indices) uses non-temporal loads so L2 stays
// reserved for the gather tables + scatter accumulators. Small tables
// (time/weather/season/month/day = 474 rows x 512 B ~ 237 KB) accumulate in
// LDS (320 KB/WGP on CDNA5) with ds_add_f32, flushed once per block.
// ---------------------------------------------------------------------------
constexpr int SMALL_ROWS = T_N + W_N + S_N + MW_N + DW_N; // 474
constexpr int R_TIME = 0;
constexpr int R_WEA  = T_N;
constexpr int R_SEA  = R_WEA + W_N;
constexpr int R_MON  = R_SEA + S_N;
constexpr int R_DAY  = R_MON + MW_N;

__global__ __launch_bounds__(512) void edge_kernel(
    const float* __restrict__ user_emb, const float* __restrict__ poi_emb,
    const float* __restrict__ time_emb, const float* __restrict__ e_W,
    const int* __restrict__ u_idx, const int* __restrict__ p_idx,
    const int* __restrict__ t_idx, const int* __restrict__ s_idx,
    const int* __restrict__ w_now, const int* __restrict__ w_day,
    const int* __restrict__ w_month, float* __restrict__ ws) {
  __shared__ float lsum[SMALL_ROWS * DD];
  __shared__ float lcnt[SMALL_ROWS];
  for (int i = threadIdx.x; i < SMALL_ROWS * DD; i += blockDim.x) lsum[i] = 0.0f;
  for (int i = threadIdx.x; i < SMALL_ROWS; i += blockDim.x) lcnt[i] = 0.0f;
  __syncthreads();

  float* user_sum = ws + WS_USER_SUM;
  float* user_cnt = ws + WS_USER_CNT;
  float* poi_sum  = ws + WS_POI_SUM;
  float* poi_cnt  = ws + WS_POI_CNT;

  const int wave = threadIdx.x >> 5;
  const int lane = threadIdx.x & 31;
  const int d4 = lane * 4;
  const int wavesPerBlock = blockDim.x >> 5;
  const int stride = gridDim.x * wavesPerBlock;

  for (int e = blockIdx.x * wavesPerBlock + wave; e < E_N; e += stride) {
    // stream-once index loads: non-temporal
    const int u  = __builtin_nontemporal_load(u_idx + e);
    const int p  = __builtin_nontemporal_load(p_idx + e);
    const int t  = __builtin_nontemporal_load(t_idx + e);
    const int s  = __builtin_nontemporal_load(s_idx + e);
    const int w  = __builtin_nontemporal_load(w_now + e);
    const int mw = __builtin_nontemporal_load(w_month + e);
    const int dy0 = __builtin_nontemporal_load(w_day + 3 * e + 0);
    const int dy1 = __builtin_nontemporal_load(w_day + 3 * e + 1);
    const int dy2 = __builtin_nontemporal_load(w_day + 3 * e + 2);

    // stream-once 512 MB e_W: non-temporal; prefetch next row for this wave
    const f4 ew = __builtin_nontemporal_load((const f4*)(e_W + (size_t)e * DD + d4));
    __builtin_prefetch(e_W + (size_t)(e + stride) * DD + d4, 0, 1);
    // gather tables: cached (L2-resident: user 5 MB, poi 25 MB, time 24 KB)
    const f4 ug = *(const f4*)(user_emb + (size_t)u * DD + d4);
    const f4 pg = *(const f4*)(poi_emb + (size_t)p * DD + d4);
    const f4 tg = *(const f4*)(time_emb + (size_t)t * DD + d4);

    // new_user message: p + t + e  -> scatter to u
    float* us = user_sum + (size_t)u * DD + d4;
    unsafeAtomicAdd(us + 0, pg.x + tg.x + ew.x);
    unsafeAtomicAdd(us + 1, pg.y + tg.y + ew.y);
    unsafeAtomicAdd(us + 2, pg.z + tg.z + ew.z);
    unsafeAtomicAdd(us + 3, pg.w + tg.w + ew.w);

    // new_poi message: u + t + e -> scatter to p
    float* ps = poi_sum + (size_t)p * DD + d4;
    unsafeAtomicAdd(ps + 0, ug.x + tg.x + ew.x);
    unsafeAtomicAdd(ps + 1, ug.y + tg.y + ew.y);
    unsafeAtomicAdd(ps + 2, ug.z + tg.z + ew.z);
    unsafeAtomicAdd(ps + 3, ug.w + tg.w + ew.w);

    // shared message u + p for the small tables (LDS, ds_add_f32)
    const float mx = ug.x + pg.x, my = ug.y + pg.y,
                mz = ug.z + pg.z, mw4 = ug.w + pg.w;
    int rows[7] = { R_TIME + t, R_WEA + w, R_SEA + s, R_MON + mw,
                    R_DAY + dy0, R_DAY + dy1, R_DAY + dy2 };
#pragma unroll
    for (int k = 0; k < 7; ++k) {
      float* lp = lsum + rows[k] * DD + d4;
      atomicAdd(lp + 0, mx);
      atomicAdd(lp + 1, my);
      atomicAdd(lp + 2, mz);
      atomicAdd(lp + 3, mw4);
    }

    if (lane == 0) {
      unsafeAtomicAdd(user_cnt + u, 1.0f);
      unsafeAtomicAdd(poi_cnt + p, 1.0f);
      atomicAdd(&lcnt[R_TIME + t], 1.0f);
      atomicAdd(&lcnt[R_WEA + w], 1.0f);
      atomicAdd(&lcnt[R_SEA + s], 1.0f);
      atomicAdd(&lcnt[R_MON + mw], 1.0f);
      atomicAdd(&lcnt[R_DAY + dy0], 1.0f);
      atomicAdd(&lcnt[R_DAY + dy1], 1.0f);
      atomicAdd(&lcnt[R_DAY + dy2], 1.0f);
    }
  }
  __syncthreads();

  // flush LDS accumulators to global
  for (int i = threadIdx.x; i < SMALL_ROWS * DD; i += blockDim.x) {
    const int row = i / DD, d = i % DD;
    float* dst;
    if (row < R_WEA)      dst = ws + WS_TIME_SUM + (size_t)(row - R_TIME) * DD + d;
    else if (row < R_SEA) dst = ws + WS_WEA_SUM + (size_t)(row - R_WEA) * DD + d;
    else if (row < R_MON) dst = ws + WS_SEA_SUM + (size_t)(row - R_SEA) * DD + d;
    else if (row < R_DAY) dst = ws + WS_MON_SUM + (size_t)(row - R_MON) * DD + d;
    else                  dst = ws + WS_DAY_SUM + (size_t)(row - R_DAY) * DD + d;
    unsafeAtomicAdd(dst, lsum[i]);
  }
  for (int i = threadIdx.x; i < SMALL_ROWS; i += blockDim.x) {
    float* dst;
    if (i < R_WEA)      dst = ws + WS_TIME_CNT + (i - R_TIME);
    else if (i < R_SEA) dst = ws + WS_WEA_CNT + (i - R_WEA);
    else if (i < R_MON) dst = ws + WS_SEA_CNT + (i - R_SEA);
    else if (i < R_DAY) dst = ws + WS_MON_CNT + (i - R_MON);
    else                dst = ws + WS_DAY_CNT + (i - R_DAY);
    unsafeAtomicAdd(dst, lcnt[i]);
  }
}

// ---------------------------------------------------------------------------
// Kernel 3: generic scatter-mean finalize: out = sum / (cnt + eps)
// ---------------------------------------------------------------------------
__global__ __launch_bounds__(128) void finalize_mean(
    const float* __restrict__ sum, const float* __restrict__ cnt,
    float* __restrict__ out) {
  const int r = blockIdx.x, d = threadIdx.x;
  out[(size_t)r * DD + d] = sum[(size_t)r * DD + d] / (cnt[r] + EPSF);
}

// ---------------------------------------------------------------------------
// Kernel 4: finalize user means -> nu (f32), nub (bf16 RNE), sq norms
// ---------------------------------------------------------------------------
__global__ __launch_bounds__(128) void finalize_user(
    const float* __restrict__ sum, const float* __restrict__ cnt,
    float* __restrict__ nu, unsigned short* __restrict__ nub,
    float* __restrict__ sq) {
  __shared__ float red[128];
  const int r = blockIdx.x, d = threadIdx.x;
  const float v = sum[(size_t)r * DD + d] / (cnt[r] + EPSF);
  nu[(size_t)r * DD + d] = v;
  // f32 -> bf16 round-to-nearest-even
  unsigned bits = __float_as_uint(v);
  unsigned rnd = bits + 0x7FFFu + ((bits >> 16) & 1u);
  nub[(size_t)r * DD + d] = (unsigned short)(rnd >> 16);
  red[d] = v * v;
  __syncthreads();
  for (int s = 64; s > 0; s >>= 1) {
    if (d < s) red[d] += red[d + s];
    __syncthreads();
  }
  if (d == 0) sq[r] = red[0];
}

// ---------------------------------------------------------------------------
// Kernel 5: kNN smoothing via bf16 WMMA Gram matrix.
// Block = 256 threads (8 waves) owns rows [iBase, iBase+16). Per 128-col slab
// each wave computes a 16x16 tile of G = X X^T with v_wmma_f32_16x16x32_bf16
// (all 4 B K-chunks loaded up front so the WMMAs issue back-to-back), then ALL
// 256 threads run top-6 selection in parallel: thread (row, subset) owns an
// 8-column slice, keeping a private top-6 across slabs; one 96-candidate merge
// per row at the end.
// ---------------------------------------------------------------------------
__global__ __launch_bounds__(256) void knn_kernel(
    const float* __restrict__ nu, const unsigned short* __restrict__ nub,
    const float* __restrict__ sq, float* __restrict__ out_user) {
  __shared__ float d2s[16 * 128];
  __shared__ float candD[16 * 96];
  __shared__ int candI[16 * 96];
  __shared__ int topIdxS[16 * 6];

  const int iBase = blockIdx.x * 16;
  const int wave = threadIdx.x >> 5;
  const int lane = threadIdx.x & 31;
  const int tid = threadIdx.x;
  const int lhalf = lane >> 4;  // 0 or 1
  const int l16 = lane & 15;

  // selection ownership: row selRow, columns [selC0, selC0+8)
  const int selRow = tid >> 4;
  const int selC0 = (tid & 15) * 8;

  // per-thread top-6 over owned column slice (across all slabs)
  float bd[6];
  int bi[6];
#pragma unroll
  for (int k = 0; k < 6; ++k) { bd[k] = 3.402823466e38f; bi[k] = 0; }

  union Frag { v16bf v; u4 q[2]; };

  // A fragments fixed per block: row = iBase + l%16,
  // K-pair layout: (v/4)*16 + (l/16)*8 + (v%4)*2  (16-bit A 16x32, ISA 7.12.2)
  const int mRow = iBase + l16;
  Frag A[4];
#pragma unroll
  for (int kc = 0; kc < 4; ++kc) {
    const unsigned short* ap = nub + (size_t)mRow * DD + kc * 32 + lhalf * 8;
    A[kc].q[0] = *(const u4*)(ap);        // v=0..3 : K = base+0..7
    A[kc].q[1] = *(const u4*)(ap + 16);   // v=4..7 : K = base+16..23
  }
  // squared norms of this block's rows, per lane (fixed across slabs)
  float sqI[8];
#pragma unroll
  for (int r = 0; r < 8; ++r) sqI[r] = sq[iBase + lhalf * 8 + r];

  for (int jBase = 0; jBase < U_N; jBase += 128) {
    const int j0 = jBase + wave * 16;
    const int nCol = min(j0 + l16, U_N - 1);  // clamp; masked in selection

    // B 32x16 bf16: lane l: col = l%16, K-pair = (v/4)*16 + (l/16)*8 + (v%4)*2.
    // Load all 4 K-chunks first (one clause of 8x b128) so the four WMMAs
    // can issue back-to-back without exposing load latency per chunk.
    Frag B[4];
#pragma unroll
    for (int kc = 0; kc < 4; ++kc) {
      const unsigned short* bp = nub + (size_t)nCol * DD + kc * 32 + lhalf * 8;
      B[kc].q[0] = *(const u4*)(bp);      // v=0..3 : K = base+0..7
      B[kc].q[1] = *(const u4*)(bp + 16); // v=4..7 : K = base+16..23
    }

    v8f acc = {0.f, 0.f, 0.f, 0.f, 0.f, 0.f, 0.f, 0.f};
#pragma unroll
    for (int kc = 0; kc < 4; ++kc) {
      acc = __builtin_amdgcn_wmma_f32_16x16x32_bf16(
          false, A[kc].v, false, B[kc].v, (short)0, acc, false, false);
    }

    // C layout: lane l holds col n = l%16, rows m = (l/16)*8 + r in VGPR r
    const float sqJ = sq[nCol];
#pragma unroll
    for (int r = 0; r < 8; ++r) {
      const int m = lhalf * 8 + r;
      d2s[m * 128 + wave * 16 + l16] = sqI[r] + sqJ - 2.0f * acc[r];
    }
    __syncthreads();

    // fully parallel selection: each thread scans its 8-column slice
    const int cmax = min(128, U_N - jBase);
    const int cend = min(selC0 + 8, cmax);
    for (int c = selC0; c < cend; ++c) {
      const float dv = d2s[selRow * 128 + c];
      int mx = 0;
#pragma unroll
      for (int k = 1; k < 6; ++k)
        if (bd[k] > bd[mx]) mx = k;
      if (dv < bd[mx]) { bd[mx] = dv; bi[mx] = jBase + c; }
    }
    __syncthreads();
  }

  // merge: 16 subsets x 6 candidates per row -> global top-6
#pragma unroll
  for (int k = 0; k < 6; ++k) {
    candD[selRow * 96 + (tid & 15) * 6 + k] = bd[k];
    candI[selRow * 96 + (tid & 15) * 6 + k] = bi[k];
  }
  __syncthreads();
  if (tid < 16) {
    float fd[6];
    int fi[6];
#pragma unroll
    for (int k = 0; k < 6; ++k) { fd[k] = 3.402823466e38f; fi[k] = 0; }
    for (int c = 0; c < 96; ++c) {
      const float dv = candD[tid * 96 + c];
      int mx = 0;
#pragma unroll
      for (int k = 1; k < 6; ++k)
        if (fd[k] > fd[mx]) mx = k;
      if (dv < fd[mx]) { fd[mx] = dv; fi[mx] = candI[tid * 96 + c]; }
    }
#pragma unroll
    for (int k = 0; k < 6; ++k) topIdxS[tid * 6 + k] = fi[k];
  }
  __syncthreads();

  // mean of the 6 neighbor embeddings (f32 source)
  for (int i = tid; i < 16 * DD; i += 256) {
    const int m = i >> 7, d = i & 127;
    float s = 0.0f;
#pragma unroll
    for (int k = 0; k < 6; ++k)
      s += nu[(size_t)topIdxS[m * 6 + k] * DD + d];
    out_user[(size_t)(iBase + m) * DD + d] = s * (1.0f / 6.0f);
  }
}

// ---------------------------------------------------------------------------
// Launch
// ---------------------------------------------------------------------------
extern "C" void kernel_launch(void* const* d_in, const int* in_sizes, int n_in,
                              void* d_out, int out_size, void* d_ws,
                              size_t ws_size, hipStream_t stream) {
  const float* user_emb = (const float*)d_in[0];
  const float* poi_emb  = (const float*)d_in[1];
  const float* time_emb = (const float*)d_in[2];
  // d_in[3..6] = weather/season/day/month embeddings: unused by the outputs
  const float* e_W      = (const float*)d_in[7];
  const int* u_idx  = (const int*)d_in[8];
  const int* p_idx  = (const int*)d_in[9];
  const int* t_idx  = (const int*)d_in[10];
  const int* s_idx  = (const int*)d_in[11];
  const int* w_now  = (const int*)d_in[12];
  const int* w_day  = (const int*)d_in[13];
  const int* w_mon  = (const int*)d_in[14];

  float* ws = (float*)d_ws;
  float* out = (float*)d_out;
  unsigned short* nub = (unsigned short*)(ws + WS_NUB);

  // 1) zero accumulators
  zero_kernel<<<2048, 256, 0, stream>>>(ws, WS_ACC_END);

  // 2) edge scatter (HBM-bound: 512 MB e_W stream + L2-resident gathers)
  edge_kernel<<<2048, 512, 0, stream>>>(user_emb, poi_emb, time_emb, e_W,
                                        u_idx, p_idx, t_idx, s_idx, w_now,
                                        w_day, w_mon, ws);

  // 3) finalize means directly into d_out for all tables except user
  finalize_mean<<<P_N, 128, 0, stream>>>(ws + WS_POI_SUM, ws + WS_POI_CNT, out + O_POI);
  finalize_mean<<<T_N, 128, 0, stream>>>(ws + WS_TIME_SUM, ws + WS_TIME_CNT, out + O_TIME);
  finalize_mean<<<W_N, 128, 0, stream>>>(ws + WS_WEA_SUM, ws + WS_WEA_CNT, out + O_WEA);
  finalize_mean<<<S_N, 128, 0, stream>>>(ws + WS_SEA_SUM, ws + WS_SEA_CNT, out + O_SEA);
  finalize_mean<<<DW_N, 128, 0, stream>>>(ws + WS_DAY_SUM, ws + WS_DAY_CNT, out + O_DAY);
  finalize_mean<<<MW_N, 128, 0, stream>>>(ws + WS_MON_SUM, ws + WS_MON_CNT, out + O_MON);

  // 4) pre-smooth user means -> nu / bf16 / squared norms
  finalize_user<<<U_N, 128, 0, stream>>>(ws + WS_USER_SUM, ws + WS_USER_CNT,
                                         ws + WS_NU, nub, ws + WS_SQ);

  // 5) kNN smoothing via WMMA Gram matrix (25.6 GFLOP, L2-resident operands)
  knn_kernel<<<U_N / 16, 256, 0, stream>>>(ws + WS_NU, nub, ws + WS_SQ,
                                           out + O_USER);
}
